// Encoder_Pos_31310311588436
// MI455X (gfx1250) — matrix operand order repaired and checked by reading, not compiled, not run on hardware
//
#include <hip/hip_runtime.h>
#include <math.h>

typedef _Float16 half_t;
typedef _Float16 v16h __attribute__((ext_vector_type(16)));
typedef _Float16 v8h  __attribute__((ext_vector_type(8)));
typedef float    v8f  __attribute__((ext_vector_type(8)));

#define EPI_BIAS  1
#define EPI_RELU  2
#define EPI_GELU  4
#define EPI_SCALE 8
#define EPI_ACCUM 16
#define EPI_BN    32
#define EPI_RES   64

// ---------------------------------------------------------------------------
// Generic batched WMMA GEMM:  C = epilogue( A(MxK) * B(KxN) )
//   TA=1 -> A stored [K][M] (strided reads), TB=1 -> B stored [N][K]
//   batch index z = z1*batch2 + z2, base += z1*s?1 + z2*s?2
// Tiles: 128x128x32, 8 waves (wave32), each wave 64x32 = 4x2 WMMA 16x16x32.
// f32 -> f16 conversion happens in the LDS staging path; accumulate f32.
// Template specialization keeps the staging path straight-line (no runtime
// trans branches in the hot loop).
// ---------------------------------------------------------------------------
template <int TA, int TB>
__global__ __launch_bounds__(256) void wmma_gemm(
    const float* __restrict__ A, int lda, long sA1, long sA2,
    const float* __restrict__ B, int ldb, long sB1, long sB2,
    float* __restrict__ C, int ldc, long sC1, long sC2,
    int M, int N, int K, int batch2,
    const float* __restrict__ bias,
    const float* __restrict__ bnScale, const float* __restrict__ bnShift,
    const float* __restrict__ res, long sR1, long sR2, int ldres,
    const float* __restrict__ alphaPtr, float alphaImm, int flags)
{
  __shared__ half_t As[2][128][32];   // 8KB per buffer
  __shared__ half_t Bs[2][128][32];   // stored transposed: [n][k]

  const int z  = blockIdx.z;
  const int z1 = z / batch2;
  const int z2 = z - z1 * batch2;
  A += z1 * sA1 + z2 * sA2;
  B += z1 * sB1 + z2 * sB2;
  C += z1 * sC1 + z2 * sC2;
  const float* resp = res ? (res + z1 * sR1 + z2 * sR2) : (const float*)0;

  const int m0  = blockIdx.y * 128;
  const int n0  = blockIdx.x * 128;
  const int tid = threadIdx.x;
  const int lane = tid & 31;
  const int wid  = tid >> 5;
  const int wM = (wid & 1) * 64;   // 2 waves along M
  const int wN = (wid >> 1) * 32;  // 4 waves along N

  v8f acc[4][2] = {};

  // staging work split (256 threads, 128x32 elements each tile)
  const int ar  = tid >> 1;          // A row 0..127
  const int ac  = (tid & 1) * 16;    // A k-base
  const int bnt = tid >> 1;          // TB=1: n index
  const int bkt = (tid & 1) * 16;    // TB=1: k-base
  const int bkn = tid >> 3;          // TB=0: k index 0..31
  const int bnn = (tid & 7) * 16;    // TB=0: n-base

  const bool ldaV = (lda & 3) == 0;
  const bool ldbV = (ldb & 3) == 0;

  auto stage = [&](int buf, int kk) {
    // ---- A tile ----
    {
      int gm = m0 + ar;
      int gk0 = kk + ac;
      if (!TA && ldaV && gm < M && gk0 + 16 <= K) {
        const float4* src = (const float4*)(A + (long)gm * lda + gk0);
        #pragma unroll
        for (int j = 0; j < 4; ++j) {
          float4 v4 = src[j];
          As[buf][ar][ac + 4 * j + 0] = (half_t)v4.x;
          As[buf][ar][ac + 4 * j + 1] = (half_t)v4.y;
          As[buf][ar][ac + 4 * j + 2] = (half_t)v4.z;
          As[buf][ar][ac + 4 * j + 3] = (half_t)v4.w;
        }
      } else {
        #pragma unroll
        for (int i = 0; i < 16; ++i) {
          int gk = gk0 + i;
          float v = 0.f;
          if (gm < M && gk < K)
            v = TA ? A[(long)gk * lda + gm] : A[(long)gm * lda + gk];
          As[buf][ar][ac + i] = (half_t)v;
        }
      }
    }
    // ---- B tile (LDS layout [n][k]) ----
    if (TB) {
      int gn = n0 + bnt;
      int gk0 = kk + bkt;
      if (ldbV && gn < N && gk0 + 16 <= K) {
        const float4* src = (const float4*)(B + (long)gn * ldb + gk0);
        #pragma unroll
        for (int j = 0; j < 4; ++j) {
          float4 v4 = src[j];
          Bs[buf][bnt][bkt + 4 * j + 0] = (half_t)v4.x;
          Bs[buf][bnt][bkt + 4 * j + 1] = (half_t)v4.y;
          Bs[buf][bnt][bkt + 4 * j + 2] = (half_t)v4.z;
          Bs[buf][bnt][bkt + 4 * j + 3] = (half_t)v4.w;
        }
      } else {
        #pragma unroll
        for (int i = 0; i < 16; ++i) {
          int gk = gk0 + i;
          float v = 0.f;
          if (gn < N && gk < K) v = B[(long)gn * ldb + gk];
          Bs[buf][bnt][bkt + i] = (half_t)v;
        }
      }
    } else {
      int gk = kk + bkn;
      int gn0 = n0 + bnn;
      if (ldbV && gk < K && gn0 + 16 <= N) {
        const float4* src = (const float4*)(B + (long)gk * ldb + gn0);
        #pragma unroll
        for (int j = 0; j < 4; ++j) {
          float4 v4 = src[j];
          Bs[buf][bnn + 4 * j + 0][bkn] = (half_t)v4.x;
          Bs[buf][bnn + 4 * j + 1][bkn] = (half_t)v4.y;
          Bs[buf][bnn + 4 * j + 2][bkn] = (half_t)v4.z;
          Bs[buf][bnn + 4 * j + 3][bkn] = (half_t)v4.w;
        }
      } else {
        #pragma unroll
        for (int i = 0; i < 16; ++i) {
          int gn = gn0 + i;
          float v = 0.f;
          if (gn < N && gk < K) v = B[(long)gk * ldb + gn];
          Bs[buf][bnn + i][bkn] = (half_t)v;
        }
      }
    }
  };

  stage(0, 0);
  __syncthreads();

  const int rsel = lane >> 4;     // 0/1 half-lane group
  const int rlo  = lane & 15;

  for (int kt = 0; kt < K; kt += 32) {
    const int cur = (kt >> 5) & 1;
    // speculative L2 prefetch two tiles ahead (global_prefetch_b8; OOB
    // translation failures are silently dropped -> no per-lane guards)
    if (kt + 64 < K) {
      __builtin_prefetch(TA ? &A[(long)(kt + 64 + ac) * lda + m0 + ar]
                            : &A[(long)(m0 + ar) * lda + kt + 64 + ac], 0, 1);
      __builtin_prefetch(TB ? &B[(long)(n0 + bnt) * ldb + kt + 64 + bkt]
                            : &B[(long)(kt + 64 + bkn) * ldb + n0 + bnn], 0, 1);
    }
    if (kt + 32 < K) stage(cur ^ 1, kt + 32);

    // ---- load fragments from LDS (ISA 16-bit A/B layouts) ----
    v16h af[4], bf[2];
    #pragma unroll
    for (int s = 0; s < 4; ++s) {
      const half_t* p = &As[cur][wM + s * 16 + rlo][rsel * 8];
      v8h lo = *(const v8h*)(p);
      v8h hi = *(const v8h*)(p + 16);
      #pragma unroll
      for (int i = 0; i < 8; ++i) { af[s][i] = lo[i]; af[s][8 + i] = hi[i]; }
    }
    #pragma unroll
    for (int t = 0; t < 2; ++t) {
      const half_t* p = &Bs[cur][wN + t * 16 + rlo][rsel * 8];
      v8h lo = *(const v8h*)(p);
      v8h hi = *(const v8h*)(p + 16);
      #pragma unroll
      for (int i = 0; i < 8; ++i) { bf[t][i] = lo[i]; bf[t][8 + i] = hi[i]; }
    }
    #pragma unroll
    for (int s = 0; s < 4; ++s)
      #pragma unroll
      for (int t = 0; t < 2; ++t)
        acc[s][t] = __builtin_amdgcn_wmma_f32_16x16x32_f16(
            false, af[s], false, bf[t], (short)0, acc[s][t], false, false);
    __syncthreads();
  }

  // ---- fused epilogue, C/D VGPR layout per ISA ----
  const float alpha = alphaImm * (alphaPtr ? alphaPtr[0] : 1.0f);
  #pragma unroll
  for (int s = 0; s < 4; ++s) {
    #pragma unroll
    for (int t = 0; t < 2; ++t) {
      #pragma unroll
      for (int i = 0; i < 8; ++i) {
        int mm = m0 + wM + s * 16 + rsel * 8 + i;
        int nn = n0 + wN + t * 16 + rlo;
        if (mm < M && nn < N) {
          float r = acc[s][t][i];
          if (flags & EPI_SCALE) r *= alpha;
          if (flags & EPI_BIAS)  r += bias[nn];
          if (flags & EPI_GELU)  r = 0.5f * r * (1.0f + erff(r * 0.70710678f));
          if (flags & EPI_BN)    r = r * bnScale[nn] + bnShift[nn];
          if (flags & EPI_RELU)  r = fmaxf(r, 0.0f);
          long idx = (long)mm * ldc + nn;
          if (flags & EPI_RES)   r += resp[(long)mm * ldres + nn];
          if (flags & EPI_ACCUM) r += C[idx];
          C[idx] = r;
        }
      }
    }
  }
}

// ---------------------------------------------------------------------------
// Elementwise / reduction helper kernels
// ---------------------------------------------------------------------------
#define GRID_STRIDE(total) \
  for (long idx = (long)blockIdx.x * blockDim.x + threadIdx.x; idx < (total); \
       idx += (long)gridDim.x * blockDim.x)

// bilinear downsample by even integer factor s == 2x2 average at centered taps
__global__ void resize_avg(const float* __restrict__ in, float* __restrict__ out,
                           int C, int s, int Hin) {
  long total = (long)4 * C * 1024;
  GRID_STRIDE(total) {
    int p = (int)(idx & 1023);
    int c = (int)((idx >> 10) % C);
    int b = (int)(idx / ((long)C * 1024));
    int y0 = (p >> 5) * s + (s >> 1) - 1;
    int x0 = (p & 31) * s + (s >> 1) - 1;
    const float* ip = in + ((long)b * C + c) * Hin * Hin;
    out[idx] = 0.25f * (ip[y0 * Hin + x0] + ip[y0 * Hin + x0 + 1] +
                        ip[(y0 + 1) * Hin + x0] + ip[(y0 + 1) * Hin + x0 + 1]);
  }
}

// im2col for 3x3 pad=1 on [4][Cin][32][32] -> [4096][Cin*9]
__global__ void im2col3x3(const float* __restrict__ in, float* __restrict__ out, int Cin) {
  long K = (long)Cin * 9;
  long total = 4096 * K;
  GRID_STRIDE(total) {
    int k  = (int)(idx % K);
    long m = idx / K;
    int c = k / 9, j = k - c * 9;
    int dy = j / 3 - 1, dx = j % 3 - 1;
    int p = (int)(m & 1023), b = (int)(m >> 10);
    int y = (p >> 5) + dy, x = (p & 31) + dx;
    float v = 0.f;
    if (y >= 0 && y < 32 && x >= 0 && x < 32)
      v = in[((long)b * Cin + c) * 1024 + y * 32 + x];
    out[idx] = v;
  }
}

__global__ void bn_fold(const float* __restrict__ g, const float* __restrict__ b,
                        const float* __restrict__ m, const float* __restrict__ v,
                        float* __restrict__ sc, float* __restrict__ sh, int C) {
  GRID_STRIDE(C) {
    float s = g[idx] * rsqrtf(v[idx] + 1e-5f);
    sc[idx] = s;
    sh[idx] = b[idx] - m[idx] * s;
  }
}

// cp[oc][h*32+w] = rel_h[oc][h] + rel_w[oc][w]
__global__ void cp_build(const float* __restrict__ rh, const float* __restrict__ rw,
                         float* __restrict__ cp) {
  GRID_STRIDE((long)64 * 1024) {
    int p = (int)(idx & 1023);
    int oc = (int)(idx >> 10);
    cp[idx] = rh[oc * 32 + (p >> 5)] + rw[oc * 32 + (p & 31)];
  }
}

__global__ __launch_bounds__(256) void softmax_rows(float* __restrict__ d, int cols, int ld) {
  __shared__ float red[256];
  float* p = d + (long)blockIdx.x * ld;
  int tid = threadIdx.x;
  float m = -3.4e38f;
  for (int i = tid; i < cols; i += 256) m = fmaxf(m, p[i]);
  red[tid] = m; __syncthreads();
  for (int s = 128; s > 0; s >>= 1) { if (tid < s) red[tid] = fmaxf(red[tid], red[tid + s]); __syncthreads(); }
  m = red[0]; __syncthreads();
  float sum = 0.f;
  for (int i = tid; i < cols; i += 256) { float e = __expf(p[i] - m); p[i] = e; sum += e; }
  red[tid] = sum; __syncthreads();
  for (int s = 128; s > 0; s >>= 1) { if (tid < s) red[tid] += red[tid + s]; __syncthreads(); }
  float inv = 1.0f / red[0];
  for (int i = tid; i < cols; i += 256) p[i] *= inv;
}

// softmax over (rowmax(a) - a)  ==  exp(rowmin - a) / sum
__global__ __launch_bounds__(256) void cab_softmax(float* __restrict__ d, int cols, int ld) {
  __shared__ float red[256];
  float* p = d + (long)blockIdx.x * ld;
  int tid = threadIdx.x;
  float mn = 3.4e38f;
  for (int i = tid; i < cols; i += 256) mn = fminf(mn, p[i]);
  red[tid] = mn; __syncthreads();
  for (int s = 128; s > 0; s >>= 1) { if (tid < s) red[tid] = fminf(red[tid], red[tid + s]); __syncthreads(); }
  mn = red[0]; __syncthreads();
  float sum = 0.f;
  for (int i = tid; i < cols; i += 256) { float e = __expf(mn - p[i]); p[i] = e; sum += e; }
  red[tid] = sum; __syncthreads();
  for (int s = 128; s > 0; s >>= 1) { if (tid < s) red[tid] += red[tid + s]; __syncthreads(); }
  float inv = 1.0f / red[0];
  for (int i = tid; i < cols; i += 256) p[i] *= inv;
}

__global__ __launch_bounds__(256) void layer_norm(float* __restrict__ out, const float* __restrict__ in,
                                                  const float* __restrict__ g, const float* __restrict__ b,
                                                  int cols, int ld) {
  __shared__ float red[256];
  const float* p = in + (long)blockIdx.x * ld;
  float* q = out + (long)blockIdx.x * ld;
  int tid = threadIdx.x;
  float s = 0.f;
  for (int i = tid; i < cols; i += 256) s += p[i];
  red[tid] = s; __syncthreads();
  for (int k = 128; k > 0; k >>= 1) { if (tid < k) red[tid] += red[tid + k]; __syncthreads(); }
  float mu = red[0] / cols; __syncthreads();
  float v = 0.f;
  for (int i = tid; i < cols; i += 256) { float d = p[i] - mu; v += d * d; }
  red[tid] = v; __syncthreads();
  for (int k = 128; k > 0; k >>= 1) { if (tid < k) red[tid] += red[tid + k]; __syncthreads(); }
  float rstd = rsqrtf(red[0] / cols + 1e-5f);
  for (int i = tid; i < cols; i += 256) q[i] = (p[i] - mu) * rstd * g[i] + b[i];
}

// t[b][0][:] = cls + pos[0];  t[b][1+p][:] = tmp[b][p][:] + pos[1+p]
__global__ void add_pos_cls(const float* __restrict__ tmp, const float* __restrict__ cls,
                            const float* __restrict__ pos, float* __restrict__ t) {
  GRID_STRIDE((long)4 * 1025 * 512) {
    int d = (int)(idx % 512);
    int n = (int)((idx / 512) % 1025);
    int b = (int)(idx / ((long)1025 * 512));
    float v = (n == 0) ? cls[d] : tmp[((long)b * 1024 + (n - 1)) * 512 + d];
    t[idx] = v + pos[n * 512 + d];
  }
}

// y12_cm[b][c][p] += t[b][1+p][c]   (y1 transpose-add into y2 buffer)
__global__ void y12_combine(float* __restrict__ y2, const float* __restrict__ t) {
  GRID_STRIDE((long)4 * 512 * 1024) {
    int p = (int)(idx & 1023);
    int c = (int)((idx >> 10) & 511);
    int b = (int)(idx >> 19);
    y2[idx] += t[((long)b * 1025 + 1 + p) * 512 + c];
  }
}

// result[b][c][p] = gamma*out_pm[b][p][c] + y3_pm[b*1024+p][c] + x[b][c][p]
__global__ void final_combine(const float* __restrict__ outpm, const float* __restrict__ y3,
                              const float* __restrict__ x, const float* __restrict__ gamma,
                              float* __restrict__ out) {
  float gm = gamma[0];
  GRID_STRIDE((long)4 * 512 * 1024) {
    int p = (int)(idx & 1023);
    int c = (int)((idx >> 10) & 511);
    int b = (int)(idx >> 19);
    long pc = ((long)b * 1024 + p) * 512 + c;
    out[idx] = gm * outpm[pc] + y3[pc] + x[idx];
  }
}

// ---------------------------------------------------------------------------
// Host-side launch graph
// ---------------------------------------------------------------------------
static inline dim3 ewgrid(long total) {
  long nb = (total + 255) / 256;
  if (nb > 16384) nb = 16384;
  return dim3((unsigned)nb);
}

#define GEMM(TA,TB, A,lda,sA1,sA2, B,ldb,sB1,sB2, Cm,ldc,sC1,sC2, M,N,K,Z,b2, bi,bs,bh, rs,sR1,sR2,ldr, aP,aI,fl) \
  wmma_gemm<TA,TB><<<dim3(((N)+127)/128, ((M)+127)/128, (Z)), dim3(256), 0, stream>>>( \
      A,lda,sA1,sA2, B,ldb,sB1,sB2, Cm,ldc,sC1,sC2, M,N,K,b2, bi,bs,bh, rs,sR1,sR2,ldr, aP,aI,fl)

extern "C" void kernel_launch(void* const* d_in, const int* in_sizes, int n_in,
                              void* d_out, int out_size, void* d_ws, size_t ws_size,
                              hipStream_t stream) {
  (void)in_sizes; (void)n_in; (void)out_size; (void)ws_size;
  const float* NUL = (const float*)0;

  // input indices (setup_inputs insertion order, recursively flattened)
  const float* x    = (const float*)d_in[0];
  const float* xs[4]= {(const float*)d_in[1], (const float*)d_in[2],
                       (const float*)d_in[3], (const float*)d_in[4]};
  // spp[i]: w,g,b,m,v at 5+5*i
  const float* qw = (const float*)d_in[25]; const float* qb = (const float*)d_in[26];
  const float* kw = (const float*)d_in[27]; const float* kb = (const float*)d_in[28];
  const float* vw = (const float*)d_in[29]; const float* vb = (const float*)d_in[30];
  const float* relh = (const float*)d_in[31]; const float* relw = (const float*)d_in[32];
  const float* gamma = (const float*)d_in[33];
  const float* projw = (const float*)d_in[34]; const float* projb = (const float*)d_in[35];
  const float* cls = (const float*)d_in[36];   const float* pos = (const float*)d_in[37];
  // vit blocks: base 38 + 10*i : qkv_w,w0,ln1_g,ln1_b,ln2_g,ln2_b,mlp_w1,mlp_b1,mlp_w2,mlp_b2
  const float* cabg = (const float*)d_in[98];
  const float* vcw = (const float*)d_in[99];  const float* vcb = (const float*)d_in[100];
  const float* vcg = (const float*)d_in[101]; const float* vcbb = (const float*)d_in[102];
  const float* vcm = (const float*)d_in[103]; const float* vcv = (const float*)d_in[104];

  // workspace carve-up (floats)
  float* ws = (float*)d_ws;
  size_t off = 0;
  auto alloc = [&](size_t n) { float* p = ws + off; off += (n + 63) & ~(size_t)63; return p; };
  float* rbuf  = alloc((size_t)4 * 256 * 1024);
  float* col   = alloc((size_t)4096 * 4608);
  float* multi = alloc((size_t)4096 * 512);
  float* bnS   = alloc(512);
  float* bnT   = alloc(512);
  float* cp    = alloc((size_t)64 * 1024);
  float* q_pm  = alloc((size_t)4096 * 64);
  float* k_pm  = alloc((size_t)4096 * 64);
  float* v_pm  = alloc((size_t)4096 * 512);
  float* outpm = alloc((size_t)4096 * 512);
  float* ttmp  = alloc((size_t)4096 * 512);
  float* t     = alloc((size_t)4 * 1025 * 512);
  float* qkv   = alloc((size_t)4 * 1025 * 1536);
  float* scores= alloc((size_t)16 * 1025 * 1040);   // ld padded to 1040 (16B align)
  float* obuf  = alloc((size_t)4 * 1025 * 512);
  float* pre1  = alloc((size_t)4 * 1025 * 512);
  float* ybuf  = alloc((size_t)4 * 1025 * 512);
  float* mlpb  = alloc((size_t)4 * 1025 * 512);
  float* pre2  = alloc((size_t)4 * 1025 * 512);
  float* aff   = alloc((size_t)4 * 512 * 512);
  float* y2    = alloc((size_t)4 * 512 * 1024);
  float* y3    = alloc((size_t)4096 * 512);

  float* result = (float*)d_out;                       // [4][512][32][32]
  float* attn   = (float*)d_out + (long)4 * 512 * 1024;// [4][1024][1024]

  // ---- SPP: resize -> im2col -> conv GEMM (BN+ReLU, accumulate) ----
  const int inch[4] = {32, 64, 128, 256};
  const int sfac[4] = {16, 8, 4, 2};
  const int hin [4] = {512, 256, 128, 64};
  for (int i = 0; i < 4; ++i) {
    const float* w = (const float*)d_in[5 + 5 * i];
    const float* g = (const float*)d_in[6 + 5 * i];
    const float* b = (const float*)d_in[7 + 5 * i];
    const float* m = (const float*)d_in[8 + 5 * i];
    const float* v = (const float*)d_in[9 + 5 * i];
    resize_avg<<<ewgrid((long)4 * inch[i] * 1024), 256, 0, stream>>>(xs[i], rbuf, inch[i], sfac[i], hin[i]);
    im2col3x3<<<ewgrid((long)4096 * inch[i] * 9), 256, 0, stream>>>(rbuf, col, inch[i]);
    bn_fold<<<2, 256, 0, stream>>>(g, b, m, v, bnS, bnT, 512);
    int Kc = inch[i] * 9;
    GEMM(0,1, col, Kc, 0L, 0L,  w, Kc, 0L, 0L,  multi, 512, 0L, 0L,
         4096, 512, Kc, 1, 1,  NUL, bnS, bnT,  NUL, 0L, 0L, 0,  NUL, 1.0f,
         EPI_BN | EPI_RELU | (i ? EPI_ACCUM : 0));
  }

  // ---- q/k/v 1x1 convs ----
  GEMM(0,1, multi, 512, 0L, 0L,  qw, 512, 0L, 0L,  q_pm, 64, 0L, 0L,
       4096, 64, 512, 1, 1,  qb, NUL, NUL,  NUL, 0L, 0L, 0,  NUL, 1.0f, EPI_BIAS);
  GEMM(1,1, x, 1024, (long)512*1024, 0L,  kw, 512, 0L, 0L,  k_pm, 64, (long)1024*64, 0L,
       1024, 64, 512, 4, 1,  kb, NUL, NUL,  NUL, 0L, 0L, 0,  NUL, 1.0f, EPI_BIAS);
  GEMM(1,1, x, 1024, (long)512*1024, 0L,  vw, 512, 0L, 0L,  v_pm, 512, (long)1024*512, 0L,
       1024, 512, 512, 4, 1,  vb, NUL, NUL,  NUL, 0L, 0L, 0,  NUL, 1.0f, EPI_BIAS);

  // ---- energy = scale*(q k^T) + q cp ; softmax -> attention (into d_out) ----
  cp_build<<<ewgrid((long)64 * 1024), 256, 0, stream>>>(relh, relw, cp);
  GEMM(0,1, q_pm, 64, (long)1024*64, 0L,  k_pm, 64, (long)1024*64, 0L,  attn, 1024, (long)1024*1024, 0L,
       1024, 1024, 64, 4, 1,  NUL, NUL, NUL,  NUL, 0L, 0L, 0,  NUL, 0.125f, EPI_SCALE);
  GEMM(0,0, q_pm, 64, (long)1024*64, 0L,  cp, 1024, 0L, 0L,  attn, 1024, (long)1024*1024, 0L,
       1024, 1024, 64, 4, 1,  NUL, NUL, NUL,  NUL, 0L, 0L, 0,  NUL, 1.0f, EPI_ACCUM);
  softmax_rows<<<4 * 1024, 256, 0, stream>>>(attn, 1024, 1024);

  // ---- out = v @ attn^T  (stored p-major: out_pm[b][p][c]) ----
  GEMM(0,0, attn, 1024, (long)1024*1024, 0L,  v_pm, 512, (long)1024*512, 0L,  outpm, 512, (long)1024*512, 0L,
       1024, 512, 1024, 4, 1,  NUL, NUL, NUL,  NUL, 0L, 0L, 0,  NUL, 1.0f, 0);

  // ---- ViT: proj + cls/pos ----
  GEMM(1,1, x, 1024, (long)512*1024, 0L,  projw, 512, 0L, 0L,  ttmp, 512, (long)1024*512, 0L,
       1024, 512, 512, 4, 1,  projb, NUL, NUL,  NUL, 0L, 0L, 0,  NUL, 1.0f, EPI_BIAS);
  add_pos_cls<<<ewgrid((long)4 * 1025 * 512), 256, 0, stream>>>(ttmp, cls, pos, t);

  const long sT = (long)1025 * 512, sQKV = (long)1025 * 1536, sSC = (long)1025 * 1040;
  for (int bi = 0; bi < 6; ++bi) {
    const float* qkvw = (const float*)d_in[38 + 10 * bi + 0];
    const float* w0   = (const float*)d_in[38 + 10 * bi + 1];
    const float* l1g  = (const float*)d_in[38 + 10 * bi + 2];
    const float* l1b  = (const float*)d_in[38 + 10 * bi + 3];
    const float* l2g  = (const float*)d_in[38 + 10 * bi + 4];
    const float* l2b  = (const float*)d_in[38 + 10 * bi + 5];
    const float* w1   = (const float*)d_in[38 + 10 * bi + 6];
    const float* b1   = (const float*)d_in[38 + 10 * bi + 7];
    const float* w2   = (const float*)d_in[38 + 10 * bi + 8];
    const float* b2   = (const float*)d_in[38 + 10 * bi + 9];

    GEMM(0,1, t, 512, sT, 0L,  qkvw, 512, 0L, 0L,  qkv, 1536, sQKV, 0L,
         1025, 1536, 512, 4, 1,  NUL, NUL, NUL,  NUL, 0L, 0L, 0,  NUL, 1.0f, 0);
    // scores[b][h] = sc * q_h k_h^T   (z = b*4 + h)
    GEMM(0,1, qkv, 1536, sQKV, 128L,  qkv + 512, 1536, sQKV, 128L,  scores, 1040, 4*sSC, sSC,
         1025, 1025, 128, 16, 4,  NUL, NUL, NUL,  NUL, 0L, 0L, 0,  NUL, 0.0883883476f, EPI_SCALE);
    softmax_rows<<<16 * 1025, 256, 0, stream>>>(scores, 1025, 1040);
    // o[b][n][h*128+d] = scores @ v_h
    GEMM(0,0, scores, 1040, 4*sSC, sSC,  qkv + 1024, 1536, sQKV, 128L,  obuf, 512, sT, 128L,
         1025, 128, 1025, 16, 4,  NUL, NUL, NUL,  NUL, 0L, 0L, 0,  NUL, 1.0f, 0);
    // pre1 = o @ w0^T + t ; y = LN1(pre1)
    GEMM(0,1, obuf, 512, sT, 0L,  w0, 512, 0L, 0L,  pre1, 512, sT, 0L,
         1025, 512, 512, 4, 1,  NUL, NUL, NUL,  t, sT, 0L, 512,  NUL, 1.0f, EPI_RES);
    layer_norm<<<4 * 1025, 256, 0, stream>>>(ybuf, pre1, l1g, l1b, 512, 512);
    // mlp
    GEMM(0,1, ybuf, 512, sT, 0L,  w1, 512, 0L, 0L,  mlpb, 512, sT, 0L,
         1025, 512, 512, 4, 1,  b1, NUL, NUL,  NUL, 0L, 0L, 0,  NUL, 1.0f, EPI_BIAS | EPI_GELU);
    GEMM(0,1, mlpb, 512, sT, 0L,  w2, 512, 0L, 0L,  pre2, 512, sT, 0L,
         1025, 512, 512, 4, 1,  b2, NUL, NUL,  ybuf, sT, 0L, 512,  NUL, 1.0f, EPI_BIAS | EPI_RES);
    layer_norm<<<4 * 1025, 256, 0, stream>>>(t, pre2, l2g, l2b, 512, 512);
  }

  // ---- CAB: aff = x x^T ; softmax(rowmax - aff) ; y2 = gamma*aff@x + x ----
  GEMM(0,1, x, 1024, (long)512*1024, 0L,  x, 1024, (long)512*1024, 0L,  aff, 512, (long)512*512, 0L,
       512, 512, 1024, 4, 1,  NUL, NUL, NUL,  NUL, 0L, 0L, 0,  NUL, 1.0f, 0);
  cab_softmax<<<4 * 512, 256, 0, stream>>>(aff, 512, 512);
  GEMM(0,0, aff, 512, (long)512*512, 0L,  x, 1024, (long)512*1024, 0L,  y2, 1024, (long)512*1024, 0L,
       512, 1024, 512, 4, 1,  NUL, NUL, NUL,  x, (long)512*1024, 0L, 1024,  cabg, 1.0f,
       EPI_SCALE | EPI_RES);

  // ---- y1 + y2, vc conv (BN+ReLU), final combine ----
  y12_combine<<<ewgrid((long)4 * 512 * 1024), 256, 0, stream>>>(y2, t);
  im2col3x3<<<ewgrid((long)4096 * 512 * 9), 256, 0, stream>>>(y2, col, 512);
  bn_fold<<<2, 256, 0, stream>>>(vcg, vcbb, vcm, vcv, bnS, bnT, 512);
  GEMM(0,1, col, 4608, 0L, 0L,  vcw, 4608, 0L, 0L,  y3, 512, 0L, 0L,
       4096, 512, 4608, 1, 1,  vcb, bnS, bnT,  NUL, 0L, 0L, 0,  NUL, 1.0f,
       EPI_BIAS | EPI_BN | EPI_RELU);
  final_combine<<<ewgrid((long)4 * 512 * 1024), 256, 0, stream>>>(outpm, y3, x, gamma, result);
}
